// RobertaLayer_21526376088233
// MI455X (gfx1250) — matrix operand order repaired and checked by reading, not compiled, not run on hardware
//
#include <hip/hip_runtime.h>

// ---------------------------------------------------------------------------
// RoBERTa layer for MI455X (gfx1250, wave32).
//  - every matmul on v_wmma_f32_16x16x32_bf16 (f32 accum)
//  - weight / K / V tiles staged by the Tensor Data Mover (tensor_load_to_lds,
//    TENSORcnt) with LDS pad insertion; activation tiles via
//    global_load_async_to_lds_b128 (ASYNCcnt); both double buffered
//  - softmax row reductions via DPP16 butterflies
// ---------------------------------------------------------------------------

typedef unsigned short u16;
typedef unsigned int   u32;
typedef __attribute__((ext_vector_type(16))) __bf16 v16bf;
typedef __attribute__((ext_vector_type(8)))  float  v8f;
typedef __attribute__((ext_vector_type(4)))  u32    u32x4;
typedef __attribute__((ext_vector_type(8)))  int    i32x8;
typedef __attribute__((ext_vector_type(4)))  int    i32x4;

union FragBF { v16bf v; u16 u[16]; u32 w[8]; };

constexpr int Bz = 4, Sq = 2048, Dm = 1024, Hh = 16, HDm = 64, Fz = 4096;
constexpr int Mrows = Bz * Sq;   // 8192

__device__ __forceinline__ u16 f2bf(float f) {
  u32 u = __float_as_uint(f);
  u32 r = u + 0x7FFFu + ((u >> 16) & 1u);   // RNE
  return (u16)(r >> 16);
}
__device__ __forceinline__ float gelu_f(float x) {
  return 0.5f * x * (1.0f + erff(x * 0.7071067811865476f));
}

// A-matrix (16x32, 16-bit): VGPR v holds K pair {(v&3)*2,(+1)} in group (v>>2)*16,
// +8 for lanes 16-31.  B-matrix (32x16): same K mapping as A.
__device__ __forceinline__ int kmapA(int v, int hl) {
  return ((v & 3) << 1) + ((v >> 2) << 4) + (hl << 3);
}
__device__ __forceinline__ int kmapB(int v, int hl) {
  return ((v & 3) << 1) + ((v >> 2) << 4) + (hl << 3);
}

// ---- DPP16 butterfly reductions (stay within 16-lane halves) ----
template <int CTRL>
__device__ __forceinline__ float dppmov(float x) {
  return __int_as_float(
      __builtin_amdgcn_update_dpp(0, __float_as_int(x), CTRL, 0xf, 0xf, true));
}
__device__ __forceinline__ float red16_max(float x) {
  x = fmaxf(x, dppmov<0xB1>(x));    // quad_perm(1,0,3,2)
  x = fmaxf(x, dppmov<0x4E>(x));    // quad_perm(2,3,0,1)
  x = fmaxf(x, dppmov<0x141>(x));   // row_half_mirror
  x = fmaxf(x, dppmov<0x140>(x));   // row_mirror
  return x;
}
__device__ __forceinline__ float red16_sum(float x) {
  x += dppmov<0xB1>(x);
  x += dppmov<0x4E>(x);
  x += dppmov<0x141>(x);
  x += dppmov<0x140>(x);
  return x;
}

// ---- CDNA5 async copy: global -> LDS, tracked by ASYNCcnt ----
__device__ __forceinline__ void async_load_b128(u32 lds_off, const u16* g) {
  asm volatile("global_load_async_to_lds_b128 %0, %1, off"
               :: "v"(lds_off), "v"(g) : "memory");
}
__device__ __forceinline__ void wait_async0() {
  asm volatile("s_wait_asynccnt 0x0" ::: "memory");
}
__device__ __forceinline__ void wait_tensor0() {
  __builtin_amdgcn_s_wait_tensorcnt(0);   // no-op for waves with TENSORcnt==0
}

// ---------------------------------------------------------------------------
// Tensor Data Mover: 2D bf16 tile (tile_d0 x tile_d1 elements) from a
// row-major tensor (row stride = stride0 elements) into LDS at lds_off,
// inserting pad_amount DWORDs of LDS padding every pad_interval DWORDs.
// Descriptor layout per CDNA5 ISA 8.3/8.4 (D# groups 0 and 1; 2D -> groups
// 2/3 zero).  Issue from ONE wave; uniform arguments.
// ---------------------------------------------------------------------------
__device__ __forceinline__ void tdm_load_2d(
    u32 lds_off, const u16* gptr, u32 tensor_d0, u32 tensor_d1, u32 stride0,
    int tile_d0, int tile_d1, int padi_code, int pada_code)
{
  const unsigned long long ga = (unsigned long long)(size_t)gptr;
  u32x4 g0;
  g0.x = 1u;                                            // count=1, user D#
  g0.y = lds_off;                                       // lds_addr (bytes)
  g0.z = (u32)(ga & 0xFFFFFFFFu);                       // global_addr[31:0]
  g0.w = (u32)((ga >> 32) & 0x1FFFFFFu) | (2u << 30);   // addr[56:32], type=2
  i32x8 g1;
  g1[0] = (1 << 16)                  // data_size = 1 (2 bytes)
        | (1 << 20)                  // pad_enable
        | (padi_code << 22)          // pad_interval
        | (pada_code << 25);         // pad_amount
  g1[1] = (int)(tensor_d0 << 16);                                   // dim0 lo16 @48
  g1[2] = (int)((tensor_d0 >> 16) & 0xFFFFu) | (int)(tensor_d1 << 16);
  g1[3] = (int)((tensor_d1 >> 16) & 0xFFFFu) | (tile_d0 << 16);     // tile_dim0
  g1[4] = tile_d1 & 0xFFFF;                                         // tile_dim1
  g1[5] = (int)stride0;                                             // dim0 stride
  g1[6] = 0;
  g1[7] = 0;
  i32x4 gz = {0, 0, 0, 0};
#if __clang_major__ >= 23
  i32x8 gz8 = {0, 0, 0, 0, 0, 0, 0, 0};
  __builtin_amdgcn_tensor_load_to_lds(g0, g1, gz, gz, gz8, 0);
#else
  __builtin_amdgcn_tensor_load_to_lds(g0, g1, gz, gz, 0);
#endif
}

// ---------------------------------------------------------------------------
// fp32 -> bf16 bulk convert, n4 = n/4
// ---------------------------------------------------------------------------
__global__ __launch_bounds__(256) void f32_to_bf16_vec(
    const float* __restrict__ s, u16* __restrict__ d, int n4)
{
  const int i = blockIdx.x * 256 + threadIdx.x;
  if (i < n4) {
    float4 f = ((const float4*)s)[i];
    ((u32*)d)[2 * i]     = (u32)f2bf(f.x) | ((u32)f2bf(f.y) << 16);
    ((u32*)d)[2 * i + 1] = (u32)f2bf(f.z) | ((u32)f2bf(f.w) << 16);
  }
}

// ---------------------------------------------------------------------------
// GEMM: C[M,N] = epilogue(A_bf16[M,K] @ W_bf16[K,N] + bias)
// Block tile 128x128, BK=32, 8 waves (4x2), wave tile 32x64 -> 8 WMMA/K-step.
// A tile: per-lane async LDS copies.  W tile: one TDM descriptor from wave 0.
// Double buffered; 1 barrier + asynccnt/tensorcnt waits per K-step.
// ---------------------------------------------------------------------------
template <bool GELU, bool OUT_BF16>
__global__ __launch_bounds__(256) void gemm_wmma(
    const u16* __restrict__ A, const u16* __restrict__ W,
    const float* __restrict__ bias, void* __restrict__ Cv,
    int M, int N, int K)
{
  constexpr int BM = 128, BN = 128, BK = 32;
  __shared__ u16 As[2][BM][BK + 8];   // row stride 80B  (16B aligned)
  __shared__ u16 Bs[2][BK][BN + 8];   // row stride 272B = 64+4 DWORD rows

  const int tid  = threadIdx.x;
  const int lane = tid & 31, wid = tid >> 5;
  const int wm = wid & 3, wn = wid >> 2;       // 4 x 2 wave grid
  const int hl = lane >> 4, lr = lane & 15;
  const int bm = blockIdx.x * BM, bn = blockIdx.y * BN;

  const int ar = tid >> 2, ac = (tid & 3) << 3;   // A: 512 16B chunks, 2/thread

  v8f acc[2][4] = {};

  auto issue = [&](int k0, int buf) {
    async_load_b128((u32)(size_t)&As[buf][ar][ac],
                    A + (size_t)(bm + ar) * K + k0 + ac);
    async_load_b128((u32)(size_t)&As[buf][ar + 64][ac],
                    A + (size_t)(bm + ar + 64) * K + k0 + ac);
    if (wid == 0) {
      // 128x32 bf16 weight tile: rows of 64 DWORDs + 4 DWORD LDS pad
      tdm_load_2d((u32)(size_t)&Bs[buf][0][0],
                  W + (size_t)k0 * N + bn,
                  (u32)N, (u32)K, (u32)N,
                  BN, BK, /*padi 64DW*/5, /*pada 4DW*/3);
    }
  };

  issue(0, 0);
  const int nk = K / BK;
  for (int kt = 0; kt < nk; ++kt) {
    const int buf = kt & 1;
    wait_async0();        // my A chunks landed
    wait_tensor0();       // wave0: W tile landed (others: no-op)
    __syncthreads();      // everyone's staging visible; prev buf free
    if (kt + 1 < nk) issue((kt + 1) * BK, buf ^ 1);

    FragBF a[2], bb[4];
#pragma unroll
    for (int i = 0; i < 2; ++i) {
      const u16* arow = &As[buf][wm * 32 + i * 16 + lr][0];
#pragma unroll
      for (int v = 0; v < 8; ++v)
        a[i].w[v] = *(const u32*)(arow + kmapA(v, hl));
    }
#pragma unroll
    for (int j = 0; j < 4; ++j) {
      const int col = wn * 64 + j * 16 + lr;
#pragma unroll
      for (int v = 0; v < 8; ++v) {
        const int kk = kmapB(v, hl);
        bb[j].u[2 * v]     = Bs[buf][kk][col];
        bb[j].u[2 * v + 1] = Bs[buf][kk + 1][col];
      }
    }
#pragma unroll
    for (int i = 0; i < 2; ++i)
#pragma unroll
      for (int j = 0; j < 4; ++j)
        acc[i][j] = __builtin_amdgcn_wmma_f32_16x16x32_bf16(
            false, a[i].v, false, bb[j].v, (short)0, acc[i][j], false, false);
  }

  // epilogue
#pragma unroll
  for (int i = 0; i < 2; ++i)
#pragma unroll
    for (int j = 0; j < 4; ++j) {
      const int col = bn + wn * 64 + j * 16 + lr;
      const float bval = bias[col];
#pragma unroll
      for (int v = 0; v < 8; ++v) {
        const int row = bm + wm * 32 + i * 16 + (hl << 3) + v;
        float val = acc[i][j][v] + bval;
        if (GELU) val = gelu_f(val);
        if (OUT_BF16) ((u16*)Cv)[(size_t)row * N + col] = f2bf(val);
        else          ((float*)Cv)[(size_t)row * N + col] = val;
      }
    }
}

// ---------------------------------------------------------------------------
// Flash attention: block = (b, h, 128 q-rows), 8 waves x 16 q-rows.
// K/V chunks (32 keys x 64) staged by TDM descriptors (wave 0), double
// buffered; online softmax with DPP16 reductions; both matmuls on WMMA.
// ---------------------------------------------------------------------------
__global__ __launch_bounds__(256) void flash_attn(
    const u16* __restrict__ Q, const u16* __restrict__ Kc,
    const u16* __restrict__ Vc, const float* __restrict__ mask,
    u16* __restrict__ Ctx)
{
  __shared__ u16 Ks[2][32][HDm + 8];   // rows = 32+4 DWORDs (TDM pad)
  __shared__ u16 Vs[2][32][HDm + 8];
  __shared__ u16 Ps[8][16][32 + 2];    // per-wave P transpose buffer

  const int tid = threadIdx.x, lane = tid & 31, wid = tid >> 5;
  const int hl = lane >> 4, lr = lane & 15;
  const int qchunk = blockIdx.x & 15;
  const int bh = blockIdx.x >> 4;
  const int b = bh >> 4, h = bh & 15;
  const int qbase = qchunk * 128 + wid * 16;

  FragBF aq[2];
  {
    const u16* qr = Q + ((size_t)b * Sq + qbase + lr) * Dm + h * HDm;
#pragma unroll
    for (int ks = 0; ks < 2; ++ks)
#pragma unroll
      for (int v = 0; v < 8; ++v)
        aq[ks].w[v] = *(const u32*)(qr + ks * 32 + kmapA(v, hl));
  }

  v8f oacc[4] = {};
  float mrow[8], lrow[8];
#pragma unroll
  for (int v = 0; v < 8; ++v) { mrow[v] = -1e30f; lrow[v] = 0.f; }

  const float scl = 0.125f;   // 1/sqrt(64)

  auto issueKV = [&](int kb, int buf) {
    if (wid == 0) {
      const size_t off = ((size_t)b * Sq + kb) * Dm + h * HDm;
      tdm_load_2d((u32)(size_t)&Ks[buf][0][0], Kc + off,
                  (u32)Dm, (u32)Sq, (u32)Dm, HDm, 32, /*padi 32DW*/4, /*pada*/3);
      tdm_load_2d((u32)(size_t)&Vs[buf][0][0], Vc + off,
                  (u32)Dm, (u32)Sq, (u32)Dm, HDm, 32, /*padi 32DW*/4, /*pada*/3);
    }
  };

  issueKV(0, 0);
  for (int kt = 0; kt < Sq / 32; ++kt) {
    const int kb = kt * 32, buf = kt & 1;
    wait_tensor0();
    __syncthreads();
    if (kb + 32 < Sq) issueKV(kb + 32, buf ^ 1);

    // ---- scores: two 16x16 tiles of QK^T ----
    v8f s[2] = {};
#pragma unroll
    for (int t = 0; t < 2; ++t)
#pragma unroll
      for (int ks = 0; ks < 2; ++ks) {
        FragBF bk;
        const u16* kr = &Ks[buf][t * 16 + lr][ks * 32];
#pragma unroll
        for (int v = 0; v < 8; ++v)
          bk.w[v] = *(const u32*)(kr + kmapB(v, hl));
        s[t] = __builtin_amdgcn_wmma_f32_16x16x32_bf16(
            false, aq[ks].v, false, bk.v, (short)0, s[t], false, false);
      }

    const float mk0 = mask[(size_t)b * Sq + kb + lr];
    const float mk1 = mask[(size_t)b * Sq + kb + 16 + lr];
#pragma unroll
    for (int v = 0; v < 8; ++v) {
      s[0][v] = s[0][v] * scl + mk0;
      s[1][v] = s[1][v] * scl + mk1;
    }

    // ---- online softmax: DPP16 row reductions ----
#pragma unroll
    for (int v = 0; v < 8; ++v) {
      const float cm = red16_max(fmaxf(s[0][v], s[1][v]));
      const float mn = fmaxf(mrow[v], cm);
      const float sc = __expf(mrow[v] - mn);
      mrow[v] = mn;
      const float p0 = __expf(s[0][v] - mn);
      const float p1 = __expf(s[1][v] - mn);
      s[0][v] = p0; s[1][v] = p1;
      lrow[v] = lrow[v] * sc + red16_sum(p0 + p1);
#pragma unroll
      for (int nt = 0; nt < 4; ++nt) oacc[nt][v] *= sc;
    }

    // ---- transpose P through per-wave LDS (D-layout -> A-layout) ----
#pragma unroll
    for (int t = 0; t < 2; ++t)
#pragma unroll
      for (int v = 0; v < 8; ++v)
        Ps[wid][(hl << 3) + v][t * 16 + lr] = f2bf(s[t][v]);
    asm volatile("s_wait_dscnt 0x0" ::: "memory");

    FragBF ap;
    {
      const u16* pr = &Ps[wid][lr][0];
#pragma unroll
      for (int v = 0; v < 8; ++v)
        ap.w[v] = *(const u32*)(pr + kmapA(v, hl));
    }

    // ---- ctx += P @ V ----
#pragma unroll
    for (int nt = 0; nt < 4; ++nt) {
      FragBF bv;
#pragma unroll
      for (int v = 0; v < 8; ++v) {
        const int kk = kmapB(v, hl);
        bv.u[2 * v]     = Vs[buf][kk][nt * 16 + lr];
        bv.u[2 * v + 1] = Vs[buf][kk + 1][nt * 16 + lr];
      }
      oacc[nt] = __builtin_amdgcn_wmma_f32_16x16x32_bf16(
          false, ap.v, false, bv.v, (short)0, oacc[nt], false, false);
    }
  }

  float inv[8];
#pragma unroll
  for (int v = 0; v < 8; ++v) inv[v] = 1.0f / lrow[v];
#pragma unroll
  for (int nt = 0; nt < 4; ++nt)
#pragma unroll
    for (int v = 0; v < 8; ++v) {
      const int m = qbase + (hl << 3) + v;
      const int c = h * HDm + nt * 16 + lr;
      Ctx[((size_t)b * Sq + m) * Dm + c] = f2bf(oacc[nt][v] * inv[v]);
    }
}

// ---------------------------------------------------------------------------
// out = LayerNorm(A + R)*g + beta ; optionally also bf16 copy for next GEMM.
// ---------------------------------------------------------------------------
__device__ __forceinline__ float block_sum256(float v) {
  __shared__ float red[8];
  v = red16_sum(v);
  v += __shfl_xor(v, 16, 32);
  const int lane = threadIdx.x & 31, wid = threadIdx.x >> 5;
  if (lane == 0) red[wid] = v;
  __syncthreads();
  float tot = 0.f;
#pragma unroll
  for (int i = 0; i < 8; ++i) tot += red[i];
  __syncthreads();
  return tot;
}

template <bool WRITE_BF>
__global__ __launch_bounds__(256) void residual_ln(
    const float* __restrict__ A, const float* __restrict__ R,
    const float* __restrict__ g, const float* __restrict__ bta,
    float* __restrict__ out, u16* __restrict__ outb)
{
  const int row = blockIdx.x, tid = threadIdx.x;
  const float* a = A + (size_t)row * Dm;
  const float* r = R + (size_t)row * Dm;
  float x[4];
  float s = 0.f;
#pragma unroll
  for (int i = 0; i < 4; ++i) { const int c = tid + i * 256; x[i] = a[c] + r[c]; s += x[i]; }
  const float mu = block_sum256(s) * (1.0f / Dm);
  float vs = 0.f;
#pragma unroll
  for (int i = 0; i < 4; ++i) { const float d = x[i] - mu; vs += d * d; }
  const float var = block_sum256(vs) * (1.0f / Dm);
  const float rstd = rsqrtf(var + 1e-12f);
#pragma unroll
  for (int i = 0; i < 4; ++i) {
    const int c = tid + i * 256;
    const float y = (x[i] - mu) * rstd * g[c] + bta[c];
    out[(size_t)row * Dm + c] = y;
    if (WRITE_BF) outb[(size_t)row * Dm + c] = f2bf(y);
  }
}

// ---------------------------------------------------------------------------
// Launch
// ---------------------------------------------------------------------------
extern "C" void kernel_launch(void* const* d_in, const int* in_sizes, int n_in,
                              void* d_out, int out_size, void* d_ws, size_t ws_size,
                              hipStream_t stream)
{
  (void)in_sizes; (void)n_in; (void)out_size; (void)ws_size;
  const float* x    = (const float*)d_in[0];
  const float* mask = (const float*)d_in[1];
  const float* Wq   = (const float*)d_in[2];  const float* bq  = (const float*)d_in[3];
  const float* Wk   = (const float*)d_in[4];  const float* bk  = (const float*)d_in[5];
  const float* Wv   = (const float*)d_in[6];  const float* bv  = (const float*)d_in[7];
  const float* Wo   = (const float*)d_in[8];  const float* bo  = (const float*)d_in[9];
  const float* g1   = (const float*)d_in[10]; const float* b1  = (const float*)d_in[11];
  const float* Wi   = (const float*)d_in[12]; const float* bi  = (const float*)d_in[13];
  const float* Wo2  = (const float*)d_in[14]; const float* bo2 = (const float*)d_in[15];
  const float* g2   = (const float*)d_in[16]; const float* b2  = (const float*)d_in[17];

  char* w = (char*)d_ws;
  const size_t nMD = (size_t)Mrows * Dm;      // 8.39M
  const size_t nDD = (size_t)Dm * Dm;         // 1.05M
  const size_t nDF = (size_t)Dm * Fz;         // 4.19M
  u16*   qb    = (u16*)w;   w += nMD * 2;
  u16*   kbuf  = (u16*)w;   w += nMD * 2;
  u16*   vbuf  = (u16*)w;   w += nMD * 2;
  u16*   ctxb  = (u16*)w;   w += nMD * 2;
  u16*   xb    = (u16*)w;   w += nMD * 2;
  u16*   attnb = (u16*)w;   w += nMD * 2;
  float* t0    = (float*)w; w += nMD * 4;
  float* attn  = (float*)w; w += nMD * 4;
  u16*   wqb   = (u16*)w;   w += nDD * 2;
  u16*   wkb   = (u16*)w;   w += nDD * 2;
  u16*   wvb   = (u16*)w;   w += nDD * 2;
  u16*   wob   = (u16*)w;   w += nDD * 2;
  u16*   wib   = (u16*)w;   w += nDF * 2;
  u16*   wo2b  = (u16*)w;   w += nDF * 2;
  u16*   hb    = (u16*)w;   // Mrows*Fz bf16 (67 MB)

  const dim3 blk(256);

  // one-time fp32 -> bf16 conversions (weights stay L2-resident as bf16)
  f32_to_bf16_vec<<<dim3((int)(nMD / 1024)), blk, 0, stream>>>(x,   xb,   (int)(nMD / 4));
  f32_to_bf16_vec<<<dim3((int)(nDD / 1024)), blk, 0, stream>>>(Wq,  wqb,  (int)(nDD / 4));
  f32_to_bf16_vec<<<dim3((int)(nDD / 1024)), blk, 0, stream>>>(Wk,  wkb,  (int)(nDD / 4));
  f32_to_bf16_vec<<<dim3((int)(nDD / 1024)), blk, 0, stream>>>(Wv,  wvb,  (int)(nDD / 4));
  f32_to_bf16_vec<<<dim3((int)(nDD / 1024)), blk, 0, stream>>>(Wo,  wob,  (int)(nDD / 4));
  f32_to_bf16_vec<<<dim3((int)(nDF / 1024)), blk, 0, stream>>>(Wi,  wib,  (int)(nDF / 4));
  f32_to_bf16_vec<<<dim3((int)(nDF / 1024)), blk, 0, stream>>>(Wo2, wo2b, (int)(nDF / 4));

  const dim3 gP(Mrows / 128, Dm / 128);   // (64, 8)
  const dim3 gF(Mrows / 128, Fz / 128);   // (64, 32)

  gemm_wmma<false, true ><<<gP, blk, 0, stream>>>(xb, wqb, bq, qb,   Mrows, Dm, Dm);
  gemm_wmma<false, true ><<<gP, blk, 0, stream>>>(xb, wkb, bk, kbuf, Mrows, Dm, Dm);
  gemm_wmma<false, true ><<<gP, blk, 0, stream>>>(xb, wvb, bv, vbuf, Mrows, Dm, Dm);

  flash_attn<<<dim3(Bz * Hh * (Sq / 128)), blk, 0, stream>>>(qb, kbuf, vbuf, mask, ctxb);

  gemm_wmma<false, false><<<gP, blk, 0, stream>>>(ctxb, wob, bo, t0, Mrows, Dm, Dm);
  residual_ln<true ><<<dim3(Mrows), blk, 0, stream>>>(t0, x, g1, b1, attn, attnb);

  gemm_wmma<true,  true ><<<gF, blk, 0, stream>>>(attnb, wib, bi, hb, Mrows, Fz, Dm);
  gemm_wmma<false, false><<<gP, blk, 0, stream>>>(hb, wo2b, bo2, t0, Mrows, Dm, Fz);
  residual_ln<false><<<dim3(Mrows), blk, 0, stream>>>(t0, attn, g2, b2, (float*)d_out, nullptr);
}
